// FixedNMP_58248346468777
// MI455X (gfx1250) — matrix operand order repaired and checked by reading, not compiled, not run on hardware
//
#include <hip/hip_runtime.h>

typedef _Float16 f16;
typedef unsigned int u32;
typedef __attribute__((ext_vector_type(16))) _Float16 v16h;
typedef __attribute__((ext_vector_type(2)))  _Float16 h2;
typedef __attribute__((ext_vector_type(8)))  float    v8f;
typedef __attribute__((address_space(3)))    f16      lds_f16;

// ---- problem dims ----
#define Bz 128
#define Nn 512
#define Ff 8
#define Hh 128
#define ROWS (Bz * Nn)      // 65536

// ---- GEMM tiling ----
#define BM 128
#define BN 128
#define BK 32
#define SA_STRIDE 34        // f16 units (32 + 2 pad) -> 17 dwords/row
#define SB_STRIDE 144       // f16 units (128 + 16 pad) -> 288B/row, 32B-aligned frags

// Toggle: 1 = global_load_async_to_lds_b128 staging (ASYNCcnt path),
//         0 = VGPR bounce staging fallback.
#define ASYNC_STAGE 1

// Copy 32 contiguous bytes global -> LDS (per-thread), f16 payload.
__device__ __forceinline__ void stage32B(f16* lds_dst, const f16* gsrc) {
#if ASYNC_STAGE
  u32 lo = (u32)(uintptr_t)(lds_f16*)lds_dst;   // generic -> LDS offset
  asm volatile("global_load_async_to_lds_b128 %0, %1, off"
               :: "v"(lo), "v"(gsrc) : "memory");
  asm volatile("global_load_async_to_lds_b128 %0, %1, off offset:16"
               :: "v"(lo + 16), "v"(gsrc) : "memory");
#else
  const u32* src = (const u32*)gsrc;
  u32* dst = (u32*)lds_dst;
#pragma unroll
  for (int q = 0; q < 8; ++q) dst[q] = src[q];
#endif
}

__device__ __forceinline__ void stage_fence() {
#if ASYNC_STAGE
  asm volatile("s_wait_asynccnt 0x0" ::: "memory");
#endif
}

// A-fragment (16x32 f16, MxK) per documented gfx1250 layout:
// lane L: m = L&15, half = L>>4 ; VGPR v holds K pair at
// kbase = (v<4?0:16) + half*8 + 2*(v&3)
__device__ __forceinline__ v16h load_afrag(const f16* sA, int lane) {
  int m = lane & 15, half = lane >> 4;
  const f16* row = sA + m * SA_STRIDE;
  v16h a;
#pragma unroll
  for (int v = 0; v < 8; ++v) {
    int kb = ((v & 4) ? 16 : 0) + half * 8 + 2 * (v & 3);
    h2 p = *(const h2*)(row + kb);
    a[2 * v]     = p.x;
    a[2 * v + 1] = p.y;
  }
  return a;
}

// B-fragment (32x16 f16, KxN): lane L: n = L&15, half = L>>4; element s holds
// K = (s<8?0:16) + half*8 + (s&7), matching the A-fragment K order.
__device__ __forceinline__ v16h load_bfrag(const f16* sB, int lane, int j) {
  int n = lane & 15, half = lane >> 4;
  const f16* col = sB + j * 16 + n;
  v16h b;
#pragma unroll
  for (int s = 0; s < 16; ++s) {
    int k = ((s & 8) ? 16 : 0) + half * 8 + (s & 7);
    b[s] = col[k * SB_STRIDE];
  }
  return b;
}

// =====================================================================
// embed layer 0: h0 = relu(jets @ W0 + b0), F=8 contraction (VALU)
// =====================================================================
__global__ __launch_bounds__(256) void embed0_kernel(
    const float* __restrict__ jets, const float* __restrict__ W0,
    const float* __restrict__ b0, f16* __restrict__ h0) {
  int idx = blockIdx.x * 256 + threadIdx.x;      // ROWS*Hh total
  int row = idx >> 7, hc = idx & 127;
  const float* jr = jets + (size_t)row * Ff;
  float v = b0[hc];
#pragma unroll
  for (int f = 0; f < Ff; ++f) v += jr[f] * W0[f * Hh + hc];
  h0[idx] = (f16)fmaxf(v, 0.f);
}

// =====================================================================
// Generic WMMA GEMM: out = act(A_f16[M x K](lda) @ W_f32[K x 128] + bias)
// Requires M % 128 == 0 (true for all call sites). Flags are template
// params so the epilogue is branch-free.
// =====================================================================
template <int RELU, int HAS_MASK, int OUT_F16>
__global__ __launch_bounds__(256) void gemm_aw_kernel(
    const f16* __restrict__ A, int lda,
    const float* __restrict__ W, int ldw,
    const float* __restrict__ bias,
    void* __restrict__ outP, int ldc,
    const float* __restrict__ rowmask,
    int M, int K) {
  __shared__ alignas(32) f16 sA[BM * SA_STRIDE];
  __shared__ alignas(32) f16 sB[BK * SB_STRIDE];

  int tid = threadIdx.x, wave = tid >> 5, lane = tid & 31;
  int nBase = blockIdx.x * BN;
  int mBase = blockIdx.y * BM;

  // per-thread staging coordinates
  int sa_r = tid >> 1, sa_kh = (tid & 1) << 4;           // A: row, 16-col half
  int sb_kr = tid >> 3, sb_seg = (tid & 7) << 4;         // W: k-row, 16-col seg

  v8f acc[8] = {};

  for (int kt = 0; kt < K; kt += BK) {
    // ---- stage A tile (f16 async copy): 128 rows x 32 cols ----
    stage32B(sA + sa_r * SA_STRIDE + sa_kh,
             A + (size_t)(mBase + sa_r) * lda + kt + sa_kh);
    // ---- stage W tile (f32 -> f16 convert): 32 rows x 128 cols ----
    {
      const float* src = W + (size_t)(kt + sb_kr) * ldw + nBase + sb_seg;
      h2* dst = (h2*)(sB + sb_kr * SB_STRIDE + sb_seg);
#pragma unroll
      for (int q = 0; q < 8; ++q) {
        h2 p;
        p.x = (f16)src[2 * q];
        p.y = (f16)src[2 * q + 1];
        dst[q] = p;
      }
    }
    stage_fence();
    __syncthreads();

    v16h a = load_afrag(sA + wave * 16 * SA_STRIDE, lane);
    v16h bf[8];
#pragma unroll
    for (int j = 0; j < 8; ++j) bf[j] = load_bfrag(sB, lane, j);
#pragma unroll
    for (int j = 0; j < 8; ++j)
      acc[j] = __builtin_amdgcn_wmma_f32_16x16x32_f16(
          false, a, false, bf[j], (short)0, acc[j], false, false);
    __syncthreads();
  }

  // ---- epilogue: C layout -> lane n = L&15, m = r + (L>>4)*8 ----
  int m0 = mBase + wave * 16 + ((lane >> 4) << 3);
  int n0 = nBase + (lane & 15);

  float bv[8];                       // bias depends only on column (j)
#pragma unroll
  for (int j = 0; j < 8; ++j) bv[j] = bias[n0 + j * 16];
  float rm[8];                       // rowmask depends only on row (r)
  if (HAS_MASK) {
#pragma unroll
    for (int r = 0; r < 8; ++r) rm[r] = rowmask[m0 + r];
  }

#pragma unroll
  for (int j = 0; j < 8; ++j) {
#pragma unroll
    for (int r = 0; r < 8; ++r) {
      float v = acc[j][r] + bv[j];
      if (RELU) v = fmaxf(v, 0.f);
      if (HAS_MASK) v *= rm[r];
      size_t idx = (size_t)(m0 + r) * ldc + n0 + j * 16;
      if (OUT_F16) ((f16*)outP)[idx] = (f16)v;
      else         ((float*)outP)[idx] = v;
    }
  }
}

// =====================================================================
// Batched WMMA GEMM (the hot one): agg[b] = dij[b](512x512,f16) @ msg[b](512x128,f16)
// =====================================================================
__global__ __launch_bounds__(256) void gemm_agg_kernel(
    const f16* __restrict__ dij, const f16* __restrict__ msg,
    f16* __restrict__ outBase, int ldc) {
  __shared__ alignas(32) f16 sA[BM * SA_STRIDE];
  __shared__ alignas(32) f16 sB[BK * SB_STRIDE];

  int tid = threadIdx.x, wave = tid >> 5, lane = tid & 31;
  int batch = blockIdx.z;
  const f16* A = dij + (size_t)batch * Nn * Nn;
  const f16* W = msg + (size_t)batch * Nn * Hh;
  f16* out = outBase + (size_t)batch * Nn * ldc;
  int mBase = blockIdx.y * BM;

  int sa_r = tid >> 1, sa_kh = (tid & 1) << 4;
  int sb_kr = tid >> 3, sb_seg = (tid & 7) << 4;

  v8f acc[8] = {};

  for (int kt = 0; kt < Nn; kt += BK) {
    stage32B(sA + sa_r * SA_STRIDE + sa_kh,
             A + (size_t)(mBase + sa_r) * Nn + kt + sa_kh);
    stage32B(sB + sb_kr * SB_STRIDE + sb_seg,
             W + (size_t)(kt + sb_kr) * Hh + sb_seg);
    // prefetch next K-slab of dij into cache (speculative)
    __builtin_prefetch(A + (size_t)(mBase + sa_r) * Nn + kt + BK + sa_kh, 0, 1);
    stage_fence();
    __syncthreads();

    v16h a = load_afrag(sA + wave * 16 * SA_STRIDE, lane);
    v16h bf[8];
#pragma unroll
    for (int j = 0; j < 8; ++j) bf[j] = load_bfrag(sB, lane, j);
#pragma unroll
    for (int j = 0; j < 8; ++j)
      acc[j] = __builtin_amdgcn_wmma_f32_16x16x32_f16(
          false, a, false, bf[j], (short)0, acc[j], false, false);
    __syncthreads();
  }

  int m0 = mBase + wave * 16 + ((lane >> 4) << 3);
  int n0 = lane & 15;
#pragma unroll
  for (int j = 0; j < 8; ++j)
#pragma unroll
    for (int r = 0; r < 8; ++r)
      out[(size_t)(m0 + r) * ldc + n0 + j * 16] = (f16)acc[j][r];
}

// =====================================================================
// adjacency: dij[b,i,:] = softmax(-d2 + mask_bias), one wave32 per row
// =====================================================================
__global__ __launch_bounds__(256) void adjacency_kernel(
    const float* __restrict__ jets, const float* __restrict__ mask,
    f16* __restrict__ dij) {
  int wave = threadIdx.x >> 5, lane = threadIdx.x & 31;
  int rowId = blockIdx.x * 8 + wave;            // 0..ROWS-1
  int b = rowId >> 9;

  const float* ji_p = jets + (size_t)rowId * Ff;
  float ji[Ff];
  float sqi = 0.f;
#pragma unroll
  for (int f = 0; f < Ff; ++f) { ji[f] = ji_p[f]; sqi += ji[f] * ji[f]; }

  const float* jb = jets + (size_t)b * Nn * Ff;
  const float* mb = mask + (size_t)b * Nn;

  float lg[16];
  float mx = -1e30f;
#pragma unroll
  for (int s = 0; s < 16; ++s) {
    int j = lane + (s << 5);
    const float* jr = jb + (size_t)j * Ff;
    float dot = 0.f, sq = 0.f;
#pragma unroll
    for (int f = 0; f < Ff; ++f) { float x = jr[f]; dot += ji[f] * x; sq += x * x; }
    float d2 = sqi + sq - 2.f * dot;
    float l = -d2 + (mb[j] > 0.f ? 0.f : -1.0e9f);
    lg[s] = l;
    mx = fmaxf(mx, l);
  }
#pragma unroll
  for (int off = 16; off > 0; off >>= 1) mx = fmaxf(mx, __shfl_xor(mx, off, 32));

  float ex[16];
  float sum = 0.f;
#pragma unroll
  for (int s = 0; s < 16; ++s) { ex[s] = __expf(lg[s] - mx); sum += ex[s]; }
#pragma unroll
  for (int off = 16; off > 0; off >>= 1) sum += __shfl_xor(sum, off, 32);
  float inv = 1.f / sum;

  f16* drow = dij + (size_t)rowId * Nn;
#pragma unroll
  for (int s = 0; s < 16; ++s) drow[lane + (s << 5)] = (f16)(ex[s] * inv);
}

// =====================================================================
// masked sum-pool over N: pooled[b,h] = sum_n h[b,n,h]*mask[b,n]  (f16 in/out)
// =====================================================================
__global__ __launch_bounds__(128) void pool_kernel(
    const f16* __restrict__ h, const float* __restrict__ mask,
    f16* __restrict__ pooled) {
  int b = blockIdx.x, hc = threadIdx.x;
  const f16* hb = h + (size_t)b * Nn * 256;     // h lives in hcat[:, :128], ldc=256
  const float* mb = mask + (size_t)b * Nn;
  float s = 0.f;
  for (int n = 0; n < Nn; ++n) s += (float)hb[(size_t)n * 256 + hc] * mb[n];
  pooled[b * Hh + hc] = (f16)s;
}

// =====================================================================
extern "C" void kernel_launch(void* const* d_in, const int* in_sizes, int n_in,
                              void* d_out, int out_size, void* d_ws, size_t ws_size,
                              hipStream_t stream) {
  const float* jets = (const float*)d_in[0];
  const float* mask = (const float*)d_in[1];
  const float* W0   = (const float*)d_in[2];
  const float* b0   = (const float*)d_in[3];
  const float* W1   = (const float*)d_in[4];
  const float* b1   = (const float*)d_in[5];
  const float* Wm   = (const float*)d_in[6];   // (3,128,128)
  const float* bm   = (const float*)d_in[7];   // (3,128)
  const float* Wu   = (const float*)d_in[8];   // (3,256,128)
  const float* bu   = (const float*)d_in[9];   // (3,128)
  const float* roW1 = (const float*)d_in[10];
  const float* rob1 = (const float*)d_in[11];
  const float* roW2 = (const float*)d_in[12];
  const float* rob2 = (const float*)d_in[13];
  float* out = (float*)d_out;

  // workspace layout (bytes); all f16 intermediates to halve HBM traffic
  char* ws = (char*)d_ws;
  const size_t dij_B   = (size_t)Bz * Nn * Nn * 2;     // 67,108,864
  const size_t hcat_B  = (size_t)ROWS * 256 * 2;       // 33,554,432
  const size_t msg_B   = (size_t)ROWS * Hh * 2;        // 16,777,216
  f16* dij    = (f16*)ws;
  f16* h0     = dij;                                   // aliased: used before dij
  f16* hcat0  = (f16*)(ws + dij_B);
  f16* hcat1  = (f16*)(ws + dij_B + hcat_B);
  f16* msg    = (f16*)(ws + dij_B + 2 * hcat_B);
  f16* pooled = (f16*)(ws + dij_B + 2 * hcat_B + msg_B);
  f16* t1     = pooled + Hh * Hh;
  (void)in_sizes; (void)n_in; (void)out_size; (void)ws_size;

  dim3 blk(256);

  // 1) embed layer 0 (VALU, K=8)
  embed0_kernel<<<ROWS * Hh / 256, blk, 0, stream>>>(jets, W0, b0, h0);

  // 2) embed layer 1: hcat0[:, :128] = relu(h0 @ W1 + b1)
  gemm_aw_kernel<1, 0, 1><<<dim3(1, ROWS / BM), blk, 0, stream>>>(
      h0, Hh, W1, Hh, b1, hcat0, 256, nullptr, ROWS, Hh);

  // 3) adjacency (overwrites h0 alias region afterwards; stream-ordered)
  adjacency_kernel<<<ROWS / 8, blk, 0, stream>>>(jets, mask, dij);

  // 4) three message-passing iterations, ping-pong hcat buffers
  for (int t = 0; t < 3; ++t) {
    f16* cur = (t & 1) ? hcat1 : hcat0;
    f16* nxt = (t & 1) ? hcat0 : hcat1;
    // msg = relu(h @ Wm[t] + bm[t])
    gemm_aw_kernel<1, 0, 1><<<dim3(1, ROWS / BM), blk, 0, stream>>>(
        cur, 256, Wm + (size_t)t * Hh * Hh, Hh, bm + t * Hh,
        msg, Hh, nullptr, ROWS, Hh);
    // agg = dij @ msg  -> cur[:, 128:256]
    gemm_agg_kernel<<<dim3(1, Nn / BM, Bz), blk, 0, stream>>>(
        dij, msg, cur + Hh, 256);
    // h' = relu(concat(h,agg) @ Wu[t] + bu[t]) * mask -> nxt[:, :128]
    gemm_aw_kernel<1, 1, 1><<<dim3(1, ROWS / BM), blk, 0, stream>>>(
        cur, 256, Wu + (size_t)t * 2 * Hh * Hh, Hh, bu + t * Hh,
        nxt, 256, mask, ROWS, 2 * Hh);
  }

  // 5) masked pool (final h in hcat1)
  pool_kernel<<<Bz, 128, 0, stream>>>(hcat1, mask, pooled);

  // 6) readout
  gemm_aw_kernel<1, 0, 1><<<dim3(1, 1), blk, 0, stream>>>(
      pooled, Hh, roW1, Hh, rob1, t1, Hh, nullptr, Bz, Hh);
  gemm_aw_kernel<0, 0, 0><<<dim3(1, 1), blk, 0, stream>>>(
      t1, Hh, roW2, Hh, rob2, out, 128, nullptr, Bz, Hh);
}
